// WaveformDecoder_77403900609051
// MI455X (gfx1250) — matrix operand order repaired and checked by reading, not compiled, not run on hardware
//
#include <hip/hip_runtime.h>
#include <math.h>

// ---------------- problem constants ----------------
#define BSZ   16
#define NG    512
#define LG    2048
#define HOP   512
#define ZD    128
#define HD    512
#define FSP   1025
#define TAR   263680
#define MROWS 8192          // BSZ*NG
#define LN10  2.30258509299f

// ---------------- vector types ----------------
typedef __attribute__((ext_vector_type(16))) __bf16 v16bf;
typedef __attribute__((ext_vector_type(8)))  __bf16 v8bf;
typedef __attribute__((ext_vector_type(4)))  __bf16 v4bf;
typedef __attribute__((ext_vector_type(8)))  float  v8f;

__device__ __forceinline__ v16bf cat8(v8bf a, v8bf b) {
  v16bf r;
#pragma unroll
  for (int i = 0; i < 8; ++i) { r[i] = a[i]; r[8 + i] = b[i]; }
  return r;
}

// =====================================================================
// GEMM: C[M,N] = act(A)[M,K] @ B[K,N] + bias, WMMA bf16 hi/lo split
// (x = hi + lo; acc += hi*hi + hi*lo + lo*hi  ->  ~fp32 accuracy).
//   BN_IN : A is pre-BN activations; apply (a-mean)*rstd*g+be, LeakyReLU.
//   MODSIG: epilogue applies 2*sigmoid(x)^ln10 + 1e-7.
// Block 256 thr = 8 waves, tile 128x64, wave tile 32x32 (2x2 WMMA).
// Staging is register-pipelined: float4 global loads for tile k+1 are
// issued before the WMMA block of tile k.
// =====================================================================
#define TM 128
#define TN 64
#define TK 32
#define LDK 40   // bf16 elems per LDS row (16B aligned rows)

template <bool BN_IN, bool MODSIG>
__global__ __launch_bounds__(256) void gemm_kernel(
    const float* __restrict__ A, const float* __restrict__ B,
    const float* __restrict__ bias,
    const float* __restrict__ mean, const float* __restrict__ rstd,
    const float* __restrict__ gamma, const float* __restrict__ beta,
    float* __restrict__ C, int K, int N) {
  __shared__ __align__(16) __bf16 Ah[TM][LDK];
  __shared__ __align__(16) __bf16 Al[TM][LDK];
  __shared__ __align__(16) __bf16 Bh[TN][LDK];
  __shared__ __align__(16) __bf16 Bl[TN][LDK];

  const int tid  = threadIdx.x;
  const int lane = tid & 31;
  const int wid  = tid >> 5;
  const int wm   = wid & 3;     // wave row block (0..3)
  const int wn   = wid >> 2;    // wave col block (0..1)
  const int m0   = blockIdx.y * TM;
  const int n0   = blockIdx.x * TN;
  const bool nvec = (N & 3) == 0;

  v8f acc[2][2] = {};

  // staging registers (tile double-buffer in regs, single LDS buffer)
  float4 aR[4], bR[2];
  float4 mR[4], sR[4], gR[4], eR[4];

  auto loadRegs = [&](int k0) {
#pragma unroll
    for (int s = 0; s < 4; ++s) {                 // A tile: 128x32 = 1024 float4
      int e4 = s * 256 + tid;
      int m  = e4 >> 3;
      int kk = k0 + ((e4 & 7) << 2);
      aR[s] = *(const float4*)&A[(size_t)(m0 + m) * K + kk];
      if (BN_IN) {
        mR[s] = *(const float4*)&mean[kk];
        sR[s] = *(const float4*)&rstd[kk];
        gR[s] = *(const float4*)&gamma[kk];
        eR[s] = *(const float4*)&beta[kk];
      }
    }
#pragma unroll
    for (int s = 0; s < 2; ++s) {                 // B tile: 32x64 = 512 float4
      int e4 = s * 256 + tid;
      int kb = k0 + (e4 >> 4);
      int gn = n0 + ((e4 & 15) << 2);
      if (nvec && gn + 3 < N) {
        bR[s] = *(const float4*)&B[(size_t)kb * N + gn];
      } else {
        const float* p = &B[(size_t)kb * N];
        float4 t;
        t.x = (gn + 0 < N) ? p[gn + 0] : 0.f;
        t.y = (gn + 1 < N) ? p[gn + 1] : 0.f;
        t.z = (gn + 2 < N) ? p[gn + 2] : 0.f;
        t.w = (gn + 3 < N) ? p[gn + 3] : 0.f;
        bR[s] = t;
      }
    }
  };

  auto storeLDS = [&]() {
#pragma unroll
    for (int s = 0; s < 4; ++s) {
      int e4 = s * 256 + tid;
      int m  = e4 >> 3;
      int kq = (e4 & 7) << 2;
      float v[4]  = {aR[s].x, aR[s].y, aR[s].z, aR[s].w};
      if (BN_IN) {
        float mm[4] = {mR[s].x, mR[s].y, mR[s].z, mR[s].w};
        float ss[4] = {sR[s].x, sR[s].y, sR[s].z, sR[s].w};
        float gg[4] = {gR[s].x, gR[s].y, gR[s].z, gR[s].w};
        float ee[4] = {eR[s].x, eR[s].y, eR[s].z, eR[s].w};
#pragma unroll
        for (int j = 0; j < 4; ++j) {
          float x = (v[j] - mm[j]) * ss[j] * gg[j] + ee[j];
          v[j] = x >= 0.f ? x : 0.2f * x;
        }
      }
      v4bf hi, lo;
#pragma unroll
      for (int j = 0; j < 4; ++j) {
        __bf16 h = (__bf16)v[j];
        hi[j] = h;
        lo[j] = (__bf16)(v[j] - (float)h);
      }
      *(v4bf*)&Ah[m][kq] = hi;
      *(v4bf*)&Al[m][kq] = lo;
    }
#pragma unroll
    for (int s = 0; s < 2; ++s) {
      int e4 = s * 256 + tid;
      int kb = e4 >> 4;
      int nq = (e4 & 15) << 2;
      float v[4] = {bR[s].x, bR[s].y, bR[s].z, bR[s].w};
      v4bf hi, lo;
#pragma unroll
      for (int j = 0; j < 4; ++j) {
        __bf16 h = (__bf16)v[j];
        hi[j] = h;
        lo[j] = (__bf16)(v[j] - (float)h);
      }
      // transposed LDS layout [n][k]
#pragma unroll
      for (int j = 0; j < 4; ++j) {
        Bh[nq + j][kb] = hi[j];
        Bl[nq + j][kb] = lo[j];
      }
    }
  };

  loadRegs(0);
  const int nK = K / TK;
  for (int ks = 0; ks < nK; ++ks) {
    storeLDS();
    __syncthreads();
    if (ks + 1 < nK) loadRegs((ks + 1) * TK);   // overlap with WMMA below

    const int r    = lane & 15;
    const int koff = (lane >> 4) * 8;    // A: K base (lanes 16-31 offset +8)
    const int kb   = (lane >> 4) * 8;    // B: K base (lanes 16-31 offset +8)

    v16bf bhi[2], blo[2];
#pragma unroll
    for (int tn = 0; tn < 2; ++tn) {
      int col = wn * 32 + tn * 16 + r;
      bhi[tn] = cat8(*(const v8bf*)&Bh[col][kb], *(const v8bf*)&Bh[col][kb + 16]);
      blo[tn] = cat8(*(const v8bf*)&Bl[col][kb], *(const v8bf*)&Bl[col][kb + 16]);
    }
#pragma unroll
    for (int tm = 0; tm < 2; ++tm) {
      int row = wm * 32 + tm * 16 + r;
      v16bf ahi = cat8(*(const v8bf*)&Ah[row][koff], *(const v8bf*)&Ah[row][koff + 16]);
      v16bf alo = cat8(*(const v8bf*)&Al[row][koff], *(const v8bf*)&Al[row][koff + 16]);
#pragma unroll
      for (int tn = 0; tn < 2; ++tn) {
        acc[tm][tn] = __builtin_amdgcn_wmma_f32_16x16x32_bf16(
            false, ahi, false, bhi[tn], (short)0, acc[tm][tn], false, false);
        acc[tm][tn] = __builtin_amdgcn_wmma_f32_16x16x32_bf16(
            false, ahi, false, blo[tn], (short)0, acc[tm][tn], false, false);
        acc[tm][tn] = __builtin_amdgcn_wmma_f32_16x16x32_bf16(
            false, alo, false, bhi[tn], (short)0, acc[tm][tn], false, false);
      }
    }
    __syncthreads();
  }

  // ---- epilogue: bias (+ mod_sigmoid), store ----
  const int rr = lane & 15;
  const int rb = (lane >> 4) * 8;
#pragma unroll
  for (int tm = 0; tm < 2; ++tm) {
#pragma unroll
    for (int tn = 0; tn < 2; ++tn) {
      int cn = n0 + wn * 32 + tn * 16 + rr;
      if (cn < N) {
        float bv = bias[cn];
        int rowbase = m0 + wm * 32 + tm * 16 + rb;
#pragma unroll
        for (int i = 0; i < 8; ++i) {
          float v = acc[tm][tn][i] + bv;
          if (MODSIG) {
            float s = 1.f / (1.f + __expf(-v));
            v = 2.f * __powf(s, LN10) + 1e-7f;
          }
          C[(size_t)(rowbase + i) * N + cn] = v;
        }
      }
    }
  }
}

// =====================================================================
// Deterministic 2-stage column stats for train-mode BatchNorm (512 cols)
// =====================================================================
__global__ __launch_bounds__(256) void colstats_partial(
    const float* __restrict__ h, float* __restrict__ ps, float* __restrict__ ps2) {
  int b = blockIdx.x;  // 128 blocks x 64 rows
  for (int c = threadIdx.x; c < HD; c += 256) {
    float s = 0.f, s2 = 0.f;
    for (int r = 0; r < 64; ++r) {
      float v = h[(size_t)(b * 64 + r) * HD + c];
      s += v; s2 += v * v;
    }
    ps[b * HD + c] = s; ps2[b * HD + c] = s2;
  }
}

__global__ __launch_bounds__(512) void colstats_final(
    const float* __restrict__ ps, const float* __restrict__ ps2,
    float* __restrict__ mean, float* __restrict__ rstd) {
  int c = threadIdx.x;
  float s = 0.f, s2 = 0.f;
  for (int p = 0; p < 128; ++p) { s += ps[p * HD + c]; s2 += ps2[p * HD + c]; }
  float m = s * (1.f / MROWS);
  float v = s2 * (1.f / MROWS) - m * m;
  mean[c] = m;
  rstd[c] = rsqrtf(v + 1e-5f);
}

// =====================================================================
// Per-grain spectral filtering: 3 LDS FFTs (Stockham radix-2, 2048 pt)
// =====================================================================
__device__ __forceinline__ float2 f2(float x, float y) { float2 r; r.x = x; r.y = y; return r; }
__device__ __forceinline__ float2 cadd(float2 a, float2 b) { return f2(a.x + b.x, a.y + b.y); }
__device__ __forceinline__ float2 csub(float2 a, float2 b) { return f2(a.x - b.x, a.y - b.y); }
__device__ __forceinline__ float2 cmulc(float2 a, float2 b) {
  return f2(a.x * b.x - a.y * b.y, a.x * b.y + a.y * b.x);
}

__device__ float2* wg_fft(float2* x, float2* y, float sgn) {
  const int tid = threadIdx.x;
  for (int p = 1; p < LG; p <<= 1) {
    __syncthreads();
    float fp = (float)p;
#pragma unroll
    for (int b = 0; b < 4; ++b) {
      int i = tid + b * 256;              // 0..1023
      int k = i & (p - 1);
      int j = ((i - k) << 1) + k;
      float ang = sgn * 3.14159265358979f * (float)k / fp;
      float s, c;
      __sincosf(ang, &s, &c);
      float2 u0 = x[i];
      float2 u1 = x[i + 1024];
      float2 w = f2(c * u1.x - s * u1.y, c * u1.y + s * u1.x);
      y[j] = cadd(u0, w);
      y[j + p] = csub(u0, w);
    }
    float2* t = x; x = y; y = t;
  }
  __syncthreads();
  return x;
}

__global__ __launch_bounds__(256) void grain_fft_kernel(
    const float* __restrict__ fc, const float* __restrict__ noise,
    float* __restrict__ audio_w) {
  __shared__ float2 bufA[LG];
  __shared__ float2 bufB[LG];
  const int row = blockIdx.x;
  const int tid = threadIdx.x;
  const float invN = 1.0f / (float)LG;

  // 1) Hermitian extension of real half-spectrum (real & even)
#pragma unroll
  for (int b = 0; b < 8; ++b) {
    int n = tid + b * 256;
    int k = (n <= 1024) ? n : (LG - n);
    bufA[n] = f2(fc[(size_t)row * FSP + k], 0.f);
  }
  float2* res = wg_fft(bufA, bufB, -1.f);          // FFT == L*irfft for real-even
  float2* other = (res == bufA) ? bufB : bufA;

  // 2) ir[n] = g[(n+1024)&2047]*hann[n]; pack z = ir + i*noise
#pragma unroll
  for (int b = 0; b < 8; ++b) {
    int n = tid + b * 256;
    int p = (n + 1024) & (LG - 1);
    float hn = 0.5f - 0.5f * __cosf(6.28318530718f * (float)n * invN);
    float irv = res[p].x * invN * hn;
    other[n] = f2(irv, noise[(size_t)row * LG + n]);
  }
  float2* Z = wg_fft(other, res, -1.f);
  float2* Yb = (Z == bufA) ? bufB : bufA;

  // 3) unpack the two Hermitian spectra, multiply
#pragma unroll
  for (int b = 0; b < 8; ++b) {
    int k = tid + b * 256;
    int km = (LG - k) & (LG - 1);
    float2 zc = Z[k];
    float2 zm = Z[km];
    float2 H  = f2(0.5f * (zc.x + zm.x), 0.5f * (zc.y - zm.y));
    float2 Nf = f2(0.5f * (zc.y + zm.y), -0.5f * (zc.x - zm.x));
    Yb[k] = cmulc(H, Nf);
  }
  float2* Y = wg_fft(Yb, Z, +1.f);                 // inverse (scale below)

  // 4) OLA window (edge-corrected hann) + store
  const int gi = row & (NG - 1);
#pragma unroll
  for (int b = 0; b < 8; ++b) {
    int n = tid + b * 256;
    float w = 0.5f - 0.5f * __cosf(6.28318530718f * (float)n * invN);
    if (gi == 0 && n < 1024) w = 1.f;
    if (gi == NG - 1 && n >= 1024) w = 1.f;
    audio_w[(size_t)row * LG + n] = Y[n].x * invN * w;
  }
}

// =====================================================================
// Overlap-add gather (deterministic; divisor computed analytically)
// =====================================================================
__global__ __launch_bounds__(256) void ola_kernel(
    const float* __restrict__ aw, float* __restrict__ outsum) {
  int t = blockIdx.x * 256 + threadIdx.x;
  if (t >= TAR) return;
  int b = blockIdx.y;
  int lo = (t >= LG) ? ((t - LG) >> 9) + 1 : 0;
  int hi = t >> 9; if (hi > NG - 1) hi = NG - 1;
  float s = 0.f;
  for (int gi = lo; gi <= hi; ++gi)
    s += aw[((size_t)(b * NG + gi)) * LG + (t - (gi << 9))];
  outsum[(size_t)b * TAR + t] = s / (float)(hi - lo + 1);
}

// =====================================================================
// Post conv (5 replicated channels fold to 1 kernel) + softsign
// =====================================================================
__global__ __launch_bounds__(256) void postconv_kernel(
    const float* __restrict__ x, const float* __restrict__ pp_w,
    const float* __restrict__ pp_b, float* __restrict__ out) {
  __shared__ float Keff[65];
  __shared__ float tile[256 + 64];
  int b = blockIdx.y;
  int t0 = blockIdx.x * 256;
  if (threadIdx.x < 65) {
    float s = 0.f;
    for (int c = 0; c < 5; ++c) s += pp_w[c * 65 + threadIdx.x];
    Keff[threadIdx.x] = s;
  }
  for (int i = threadIdx.x; i < 320; i += 256) {
    int idx = t0 - 32 + i;
    tile[i] = (idx >= 0 && idx < TAR) ? x[(size_t)b * TAR + idx] : 0.f;
  }
  __syncthreads();
  int t = t0 + threadIdx.x;
  if (t >= TAR) return;
  float acc = pp_b[0];
#pragma unroll
  for (int j = 0; j < 65; ++j) acc += tile[threadIdx.x + j] * Keff[j];
  out[(size_t)b * TAR + t] = acc / (1.f + fabsf(acc));
}

// =====================================================================
extern "C" void kernel_launch(void* const* d_in, const int* in_sizes, int n_in,
                              void* d_out, int out_size, void* d_ws, size_t ws_size,
                              hipStream_t stream) {
  (void)in_sizes; (void)n_in; (void)out_size; (void)ws_size;
  const float* z    = (const float*)d_in[0];
  const float* w0   = (const float*)d_in[1];
  const float* b0   = (const float*)d_in[2];
  const float* g0   = (const float*)d_in[3];
  const float* be0  = (const float*)d_in[4];
  const float* w1   = (const float*)d_in[5];
  const float* b1   = (const float*)d_in[6];
  const float* g1   = (const float*)d_in[7];
  const float* be1  = (const float*)d_in[8];
  const float* w2   = (const float*)d_in[9];
  const float* b2   = (const float*)d_in[10];
  const float* g2   = (const float*)d_in[11];
  const float* be2  = (const float*)d_in[12];
  const float* w3   = (const float*)d_in[13];
  const float* b3   = (const float*)d_in[14];
  const float* pp_w = (const float*)d_in[15];
  const float* pp_b = (const float*)d_in[16];
  const float* noise= (const float*)d_in[17];

  char* ws = (char*)d_ws;
  // h ping-pong buffers are dead before AUDIO is written -> alias region [0,64MiB)
  float* HA    = (float*)(ws);                                 // 8192*512   f32
  float* HB    = (float*)(ws + 16777216ull);                   // 8192*512   f32
  float* AUDIO = (float*)(ws);                                 // 8192*2048  f32
  float* FC    = (float*)(ws + 67108864ull);                   // 8192*1025  f32
  float* SUM   = (float*)(ws + 100696064ull);                  // 16*263680  f32
  float* PS    = (float*)(ws + 117571584ull);                  // 128*512
  float* PS2   = PS + 128 * HD;
  float* MEAN  = PS2 + 128 * HD;
  float* RSTD  = MEAN + HD;

  dim3 blk(256);
  dim3 gG(HD / TN, MROWS / TM);               // (8,64)
  dim3 gG3((FSP + TN - 1) / TN, MROWS / TM);  // (17,64)

  // layer 0: z @ w0 + b0
  gemm_kernel<false, false><<<gG, blk, 0, stream>>>(
      z, w0, b0, nullptr, nullptr, nullptr, nullptr, HA, ZD, HD);
  colstats_partial<<<128, blk, 0, stream>>>(HA, PS, PS2);
  colstats_final<<<1, 512, 0, stream>>>(PS, PS2, MEAN, RSTD);

  // layer 1: BN0+LReLU(HA) @ w1 + b1
  gemm_kernel<true, false><<<gG, blk, 0, stream>>>(
      HA, w1, b1, MEAN, RSTD, g0, be0, HB, HD, HD);
  colstats_partial<<<128, blk, 0, stream>>>(HB, PS, PS2);
  colstats_final<<<1, 512, 0, stream>>>(PS, PS2, MEAN, RSTD);

  // layer 2: BN1+LReLU(HB) @ w2 + b2
  gemm_kernel<true, false><<<gG, blk, 0, stream>>>(
      HB, w2, b2, MEAN, RSTD, g1, be1, HA, HD, HD);
  colstats_partial<<<128, blk, 0, stream>>>(HA, PS, PS2);
  colstats_final<<<1, 512, 0, stream>>>(PS, PS2, MEAN, RSTD);

  // layer 3: BN2+LReLU(HA) @ w3 + b3 -> mod_sigmoid -> FC
  gemm_kernel<true, true><<<gG3, blk, 0, stream>>>(
      HA, w3, b3, MEAN, RSTD, g2, be2, FC, HD, FSP);

  // spectral noise filtering per grain
  grain_fft_kernel<<<MROWS, blk, 0, stream>>>(FC, noise, AUDIO);

  // overlap-add + normalize
  ola_kernel<<<dim3((TAR + 255) / 256, BSZ), blk, 0, stream>>>(AUDIO, SUM);

  // post conv + softsign
  postconv_kernel<<<dim3((TAR + 255) / 256, BSZ), blk, 0, stream>>>(
      SUM, pp_w, pp_b, (float*)d_out);
}
